// GRU_37228776521967
// MI455X (gfx1250) — matrix operand order repaired and checked by reading, not compiled, not run on hardware
//
#include <hip/hip_runtime.h>
#include <hip/hip_bf16.h>

// ---------------------------------------------------------------------------
// GRU LM forward on gfx1250 (MI455X): bf16 WMMA + TDM-staged activations.
//   SEQ=64, BATCH=64, EMB=HID=1024, VOCAB=10000, LAYERS=2
// ---------------------------------------------------------------------------

typedef __attribute__((ext_vector_type(16))) __bf16 bf16x16;
typedef __attribute__((ext_vector_type(8)))  float  f32x8;
typedef __attribute__((ext_vector_type(4)))  unsigned int u32x4;
typedef __attribute__((ext_vector_type(8)))  int i32x8;
typedef __attribute__((ext_vector_type(4)))  int i32x4;

#define WMMA_BF16(a, b, c) \
  __builtin_amdgcn_wmma_f32_16x16x32_bf16(false, (a), false, (b), (short)0, (c), false, false)

static constexpr int S = 64;       // seq
static constexpr int B = 64;       // batch
static constexpr int E = 1024;     // emb
static constexpr int H = 1024;     // hid
static constexpr int V = 10000;    // vocab
static constexpr int SB = S * B;   // 4096
static constexpr int KD = 1024;    // K dim of every GEMM in this model

__device__ __forceinline__ float sigmoidf_(float x) { return 1.0f / (1.0f + __expf(-x)); }

// ---------------------------------------------------------------------------
// TDM: 1D contiguous copy of `nelem` bf16 elements global -> LDS.
// D# per CDNA5 ISA ch.8: group0 = {count, lds_addr, global_addr, type=2};
// group1 = {data_size=2B, tensor_dim0=nelem, tensor_dim1=1, tile_dim0=nelem,
// tile_dim1=1, tensor_dim0_stride=nelem}; groups 2/3 unused (zero).
// Issue from one wave only; EXEC ignored by TDM; tracked via TENSORcnt.
// This toolchain's builtin is the 6-arg form:
//   (u32x4 g0, i32x8 g1, i32x4 g2, i32x4 g3, i32x8 pad, i32 cpol)
// ---------------------------------------------------------------------------
__device__ __forceinline__ void tdm_load_1d(__bf16* lds_dst, const __bf16* gsrc, int nelem) {
  unsigned long long ga = (unsigned long long)(uintptr_t)gsrc;
  unsigned int lds = (unsigned int)(uintptr_t)lds_dst;   // low 32 bits = LDS byte address
  u32x4 g0;
  g0[0] = 1u;                                            // count=1, user descriptor
  g0[1] = lds;                                           // lds_addr
  g0[2] = (unsigned int)(ga & 0xffffffffu);              // global_addr[31:0]
  g0[3] = (unsigned int)((ga >> 32) & 0x1ffffffu) | (2u << 30);  // addr[56:32] | type=2
  i32x8 g1;
  g1[0] = 0x00010000;                                    // data_size=1 (2 bytes)
  g1[1] = (nelem & 0xffff) << 16;                        // tensor_dim0[15:0]
  g1[2] = ((nelem >> 16) & 0xffff) | (1 << 16);          // tensor_dim0[31:16] | tensor_dim1=1
  g1[3] = (nelem & 0xffff) << 16;                        // tile_dim0
  g1[4] = 1;                                             // tile_dim1=1
  g1[5] = nelem;                                         // tensor_dim0_stride[31:0]
  g1[6] = 0;
  g1[7] = 0;
  i32x4 z4 = {0, 0, 0, 0};
  i32x8 z8 = {0, 0, 0, 0, 0, 0, 0, 0};
  __builtin_amdgcn_tensor_load_to_lds(g0, g1, z4, z4, z8, 0);
}

// A-matrix fragment (16x32 bf16, MxK) from an LDS strip [16][KD]:
// lanes 0-15 hold M=lane, K in {kc..kc+7, kc+16..kc+23};
// lanes 16-31 hold M=lane-16, K in {kc+8..kc+15, kc+24..kc+31}. Two 16B ds loads.
__device__ __forceinline__ bf16x16 frag_a_lds(const __bf16* As, int kc) {
  int lane = threadIdx.x & 31;
  const __bf16* q = As + (lane & 15) * KD + kc + ((lane & 16) ? 8 : 0);
  union { bf16x16 v; float4 f[2]; } u;
  u.f[0] = *(const float4*)(q);
  u.f[1] = *(const float4*)(q + 16);
  return u.v;
}

// Same A layout, straight from global (used only by the small tail kernel).
__device__ __forceinline__ bf16x16 frag_a_glb(const __bf16* __restrict__ p, int kc) {
  int lane = threadIdx.x & 31;
  const __bf16* q = p + (size_t)(lane & 15) * KD + kc + ((lane & 16) ? 8 : 0);
  union { bf16x16 v; float4 f[2]; } u;
  u.f[0] = *(const float4*)(q);
  u.f[1] = *(const float4*)(q + 16);
  return u.v;
}

// B-matrix fragment (32x16 bf16, KxN): lane n (0-15) holds column n, K in {kc..kc+7,
// kc+16..kc+23}; lanes 16-31 hold K in {kc+8..kc+15, kc+24..kc+31}. W is [N,K] row-major, so
// column n of B == row n of W. Two 16B global loads.
__device__ __forceinline__ bf16x16 frag_b_glb(const __bf16* __restrict__ p, int kc) {
  int lane = threadIdx.x & 31;
  const __bf16* q = p + (size_t)(lane & 15) * KD + kc + ((lane & 16) ? 8 : 0);
  union { bf16x16 v; float4 f[2]; } u;
  u.f[0] = *(const float4*)(q);
  u.f[1] = *(const float4*)(q + 16);
  return u.v;
}

// cooperative copy of `rows` x KD bf16 into LDS (blockDim = 256)
__device__ __forceinline__ void stage(__bf16* dst, const __bf16* __restrict__ src, int rows) {
  const float4* s = (const float4*)src;
  float4* d = (float4*)dst;
  int n = rows * KD / 8;
  for (int i = threadIdx.x; i < n; i += 256) d[i] = s[i];
}

// ---------------------------------------------------------------------------
// setup kernels
// ---------------------------------------------------------------------------
__global__ void cvt_kernel(const float* __restrict__ s, __bf16* __restrict__ d, size_t n) {
  size_t i = (size_t)blockIdx.x * blockDim.x + threadIdx.x;
  size_t stride = (size_t)gridDim.x * blockDim.x;
  for (; i < n; i += stride) d[i] = (__bf16)s[i];
}

__global__ void embed_kernel(const int* __restrict__ idx, const float* __restrict__ emb,
                             __bf16* __restrict__ Xbf) {
  int row = blockIdx.x;                        // t*B + b, 0..4095
  const float* src = emb + (size_t)idx[row] * E;
  __bf16* dst = Xbf + (size_t)row * E;
  for (int c = threadIdx.x; c < E; c += blockDim.x) dst[c] = (__bf16)src[c];
}

__global__ void inith_kernel(const float* __restrict__ hidden,
                             float* __restrict__ h0f, __bf16* __restrict__ h0bf,
                             float* __restrict__ h1f, __bf16* __restrict__ h1bf) {
  int i = blockIdx.x * blockDim.x + threadIdx.x;
  if (i < B * H) {
    float a = hidden[i];
    float b = hidden[B * H + i];
    h0f[i] = a; h0bf[i] = (__bf16)a;
    h1f[i] = b; h1bf[i] = (__bf16)b;
  }
}

__global__ void finalh_kernel(const float* __restrict__ h0f, const float* __restrict__ h1f,
                              float* __restrict__ out) {
  int i = blockIdx.x * blockDim.x + threadIdx.x;
  if (i < B * H) { out[i] = h0f[i]; out[B * H + i] = h1f[i]; }
}

// R0 = sigmoid(Ar_t) * h0   (layer-0 reset gate needs no step-time GEMM)
__global__ void r0_kernel(const float* __restrict__ Ar_t, const float* __restrict__ h0f,
                          __bf16* __restrict__ R0bf) {
  int i = blockIdx.x * blockDim.x + threadIdx.x;
  if (i < B * H) R0bf[i] = (__bf16)(sigmoidf_(Ar_t[i]) * h0f[i]);
}

// ---------------------------------------------------------------------------
// Big bf16 GEMM: C[M,N] = A[M,KD] * W[N,KD]^T + bias, FULL 64-col groups only.
// grid = (ceil(ngroups_full/8), M/(16*MT)); block = 256 (8 waves);
// one wave = (16*MT) x 64 tile; branch-free inner loop, WMMA with full EXEC.
// ---------------------------------------------------------------------------
template <int MT>
__global__ void gemm_bias_kernel(const __bf16* __restrict__ A, const __bf16* __restrict__ W,
                                 const float* __restrict__ bias, float* __restrict__ C,
                                 int N, int ngroups_full) {
  extern __shared__ char smem[];
  __bf16* As = (__bf16*)smem;                  // (16*MT) x KD strip
  int m0 = blockIdx.y * (16 * MT);
  stage(As, A + (size_t)m0 * KD, 16 * MT);
  __syncthreads();

  int wave = __builtin_amdgcn_readfirstlane(threadIdx.x >> 5);
  int lane = threadIdx.x & 31;
  int ng = blockIdx.x * 8 + wave;              // uniform
  if (ng >= ngroups_full) return;
  const __bf16* Wp = W + (size_t)ng * 64 * KD;

  f32x8 acc[MT][4] = {};
  for (int kc = 0; kc < KD; kc += 32) {
#pragma unroll
    for (int j = 0; j < 4; ++j)
      __builtin_prefetch(Wp + (size_t)j * 16 * KD + kc + 512, 0, 1);
    bf16x16 a[MT];
#pragma unroll
    for (int mi = 0; mi < MT; ++mi) a[mi] = frag_a_lds(As + mi * 16 * KD, kc);
#pragma unroll
    for (int j = 0; j < 4; ++j) {
      bf16x16 b = frag_b_glb(Wp + (size_t)j * 16 * KD, kc);
#pragma unroll
      for (int mi = 0; mi < MT; ++mi)
        acc[mi][j] = WMMA_BF16(a[mi], b, acc[mi][j]);
    }
  }
  int cl = lane & 15;
#pragma unroll
  for (int mi = 0; mi < MT; ++mi) {
    int rb = m0 + mi * 16 + ((lane >> 4) << 3);
#pragma unroll
    for (int j = 0; j < 4; ++j) {
      int col = ng * 64 + j * 16 + cl;
      float bv = bias[col];
#pragma unroll
      for (int r = 0; r < 8; ++r)
        C[(size_t)(rb + r) * N + col] = acc[mi][j][r] + bv;
    }
  }
}

// Ragged-edge tail: one 16x16 tile per wave at fixed column base n0.
// grid = M/16/8 blocks of 8 waves; branch-free.
__global__ void gemm_bias_tail_kernel(const __bf16* __restrict__ A, const __bf16* __restrict__ W,
                                      const float* __restrict__ bias, float* __restrict__ C,
                                      int N, int n0) {
  int wave = __builtin_amdgcn_readfirstlane(threadIdx.x >> 5);
  int lane = threadIdx.x & 31;
  int m0 = (blockIdx.x * 8 + wave) * 16;
  const __bf16* Ap = A + (size_t)m0 * KD;
  const __bf16* Wp = W + (size_t)n0 * KD;

  f32x8 acc = {};
  for (int kc = 0; kc < KD; kc += 32) {
    bf16x16 a = frag_a_glb(Ap, kc);
    bf16x16 b = frag_b_glb(Wp, kc);
    acc = WMMA_BF16(a, b, acc);
  }
  int cl = lane & 15;
  int rb = m0 + ((lane >> 4) << 3);
  int col = n0 + cl;
  float bv = bias[col];
#pragma unroll
  for (int r = 0; r < 8; ++r)
    C[(size_t)(rb + r) * N + col] = acc[r] + bv;
}

// ---------------------------------------------------------------------------
// Per-step kernels: M=64, N=KD=1024; grid = (2, 4), block = 256.
// A strips are staged into LDS by the Tensor Data Mover (wave 0 issues,
// TENSORcnt waited, then block barrier publishes LDS).
// ---------------------------------------------------------------------------

// layer-0 update: z = sig(Az + Uz*h0 + buz); ht = tanh(Ah + Uh*R0 + buh); h' = (1-z)ht + z h0
__global__ void step0_kernel(const float* __restrict__ Az_t, const float* __restrict__ Ah_t,
                             const float* __restrict__ h0f, const __bf16* __restrict__ h0bf,
                             const __bf16* __restrict__ R0bf,
                             const __bf16* __restrict__ Uz, const __bf16* __restrict__ Uh,
                             const float* __restrict__ buz, const float* __restrict__ buh,
                             float* __restrict__ h0f_o, __bf16* __restrict__ h0bf_o) {
  extern __shared__ char smem[];
  __bf16* Hs = (__bf16*)smem;                  // h0 strip
  __bf16* Rs = Hs + 16 * KD;                   // R0 strip
  int m0 = blockIdx.y * 16;
  if (threadIdx.x < 32) {                      // wave 0 drives the TDM
    tdm_load_1d(Hs, h0bf + (size_t)m0 * KD, 16 * KD);
    tdm_load_1d(Rs, R0bf + (size_t)m0 * KD, 16 * KD);
    __builtin_amdgcn_s_wait_tensorcnt(0);
  }
  __syncthreads();

  int wave = __builtin_amdgcn_readfirstlane(threadIdx.x >> 5);
  int lane = threadIdx.x & 31;
  int ng = blockIdx.x * 8 + wave;              // 0..15
  const __bf16* Uzp = Uz + (size_t)ng * 64 * KD;
  const __bf16* Uhp = Uh + (size_t)ng * 64 * KD;

  f32x8 accZ[4] = {}, accH[4] = {};
  for (int kc = 0; kc < KD; kc += 32) {
    bf16x16 ah = frag_a_lds(Hs, kc);
    bf16x16 ar = frag_a_lds(Rs, kc);
#pragma unroll
    for (int j = 0; j < 4; ++j) {
      __builtin_prefetch(Uzp + (size_t)j * 16 * KD + kc + 512, 0, 1);
      __builtin_prefetch(Uhp + (size_t)j * 16 * KD + kc + 512, 0, 1);
      bf16x16 bz = frag_b_glb(Uzp + (size_t)j * 16 * KD, kc);
      bf16x16 bh = frag_b_glb(Uhp + (size_t)j * 16 * KD, kc);
      accZ[j] = WMMA_BF16(ah, bz, accZ[j]);
      accH[j] = WMMA_BF16(ar, bh, accH[j]);
    }
  }
  int cl = lane & 15;
  int rb = m0 + ((lane >> 4) << 3);
#pragma unroll
  for (int j = 0; j < 4; ++j) {
    int col = ng * 64 + j * 16 + cl;
#pragma unroll
    for (int r = 0; r < 8; ++r) {
      size_t o = (size_t)(rb + r) * H + col;
      float z  = sigmoidf_(Az_t[o] + accZ[j][r] + buz[col]);
      float ht = tanhf(Ah_t[o] + accH[j][r] + buh[col]);
      float hn = (1.0f - z) * ht + z * h0f[o];
      h0f_o[o] = hn;
      h0bf_o[o] = (__bf16)hn;
    }
  }
}

// layer-1 reset: R1 = sigmoid(Wr1*x + br1) * h1
__global__ void l1r_kernel(const __bf16* __restrict__ xbf,
                           const __bf16* __restrict__ Wr, const float* __restrict__ br,
                           const float* __restrict__ h1f, __bf16* __restrict__ R1bf) {
  extern __shared__ char smem[];
  __bf16* Xs = (__bf16*)smem;
  int m0 = blockIdx.y * 16;
  if (threadIdx.x < 32) {
    tdm_load_1d(Xs, xbf + (size_t)m0 * KD, 16 * KD);
    __builtin_amdgcn_s_wait_tensorcnt(0);
  }
  __syncthreads();

  int wave = __builtin_amdgcn_readfirstlane(threadIdx.x >> 5);
  int lane = threadIdx.x & 31;
  int ng = blockIdx.x * 8 + wave;
  const __bf16* Wp = Wr + (size_t)ng * 64 * KD;

  f32x8 acc[4] = {};
  for (int kc = 0; kc < KD; kc += 32) {
    bf16x16 a = frag_a_lds(Xs, kc);
#pragma unroll
    for (int j = 0; j < 4; ++j) {
      __builtin_prefetch(Wp + (size_t)j * 16 * KD + kc + 512, 0, 1);
      bf16x16 b = frag_b_glb(Wp + (size_t)j * 16 * KD, kc);
      acc[j] = WMMA_BF16(a, b, acc[j]);
    }
  }
  int cl = lane & 15;
  int rb = m0 + ((lane >> 4) << 3);
#pragma unroll
  for (int j = 0; j < 4; ++j) {
    int col = ng * 64 + j * 16 + cl;
#pragma unroll
    for (int r = 0; r < 8; ++r) {
      size_t o = (size_t)(rb + r) * H + col;
      float rr = sigmoidf_(acc[j][r] + br[col]);
      R1bf[o] = (__bf16)(rr * h1f[o]);
    }
  }
}

// layer-1 update: accZ = Wz1*x + Uz1*h1 ; accH = Wh1*x + Uh1*R1 (shared accumulators)
__global__ void step1_kernel(const __bf16* __restrict__ xbf, const __bf16* __restrict__ h1bf,
                             const __bf16* __restrict__ R1bf,
                             const __bf16* __restrict__ Wz, const __bf16* __restrict__ Wh,
                             const __bf16* __restrict__ Uz, const __bf16* __restrict__ Uh,
                             const float* __restrict__ bz, const float* __restrict__ buz,
                             const float* __restrict__ bh, const float* __restrict__ buh,
                             const float* __restrict__ h1f,
                             float* __restrict__ h1f_o, __bf16* __restrict__ h1bf_o,
                             __bf16* __restrict__ H1row) {
  extern __shared__ char smem[];
  __bf16* Xs = (__bf16*)smem;
  __bf16* Hs = Xs + 16 * KD;
  __bf16* Rs = Hs + 16 * KD;
  int m0 = blockIdx.y * 16;
  if (threadIdx.x < 32) {
    tdm_load_1d(Xs, xbf  + (size_t)m0 * KD, 16 * KD);
    tdm_load_1d(Hs, h1bf + (size_t)m0 * KD, 16 * KD);
    tdm_load_1d(Rs, R1bf + (size_t)m0 * KD, 16 * KD);
    __builtin_amdgcn_s_wait_tensorcnt(0);
  }
  __syncthreads();

  int wave = __builtin_amdgcn_readfirstlane(threadIdx.x >> 5);
  int lane = threadIdx.x & 31;
  int ng = blockIdx.x * 8 + wave;
  const __bf16* Wzp = Wz + (size_t)ng * 64 * KD;
  const __bf16* Whp = Wh + (size_t)ng * 64 * KD;
  const __bf16* Uzp = Uz + (size_t)ng * 64 * KD;
  const __bf16* Uhp = Uh + (size_t)ng * 64 * KD;

  f32x8 accZ[4] = {}, accH[4] = {};
  for (int kc = 0; kc < KD; kc += 32) {
    bf16x16 ax = frag_a_lds(Xs, kc);
    bf16x16 ah = frag_a_lds(Hs, kc);
    bf16x16 ar = frag_a_lds(Rs, kc);
#pragma unroll
    for (int j = 0; j < 4; ++j) {
      __builtin_prefetch(Wzp + (size_t)j * 16 * KD + kc + 512, 0, 1);
      __builtin_prefetch(Uzp + (size_t)j * 16 * KD + kc + 512, 0, 1);
      __builtin_prefetch(Whp + (size_t)j * 16 * KD + kc + 512, 0, 1);
      __builtin_prefetch(Uhp + (size_t)j * 16 * KD + kc + 512, 0, 1);
      bf16x16 bwz = frag_b_glb(Wzp + (size_t)j * 16 * KD, kc);
      bf16x16 buzf = frag_b_glb(Uzp + (size_t)j * 16 * KD, kc);
      bf16x16 bwh = frag_b_glb(Whp + (size_t)j * 16 * KD, kc);
      bf16x16 buhf = frag_b_glb(Uhp + (size_t)j * 16 * KD, kc);
      accZ[j] = WMMA_BF16(ax, bwz, accZ[j]);
      accZ[j] = WMMA_BF16(ah, buzf, accZ[j]);
      accH[j] = WMMA_BF16(ax, bwh, accH[j]);
      accH[j] = WMMA_BF16(ar, buhf, accH[j]);
    }
  }
  int cl = lane & 15;
  int rb = m0 + ((lane >> 4) << 3);
#pragma unroll
  for (int j = 0; j < 4; ++j) {
    int col = ng * 64 + j * 16 + cl;
#pragma unroll
    for (int r = 0; r < 8; ++r) {
      size_t o = (size_t)(rb + r) * H + col;
      float z  = sigmoidf_(accZ[j][r] + bz[col] + buz[col]);
      float ht = tanhf(accH[j][r] + bh[col] + buh[col]);
      float hn = (1.0f - z) * ht + z * h1f[o];
      h1f_o[o] = hn;
      h1bf_o[o] = (__bf16)hn;
      H1row[o] = (__bf16)hn;       // feeds final logits GEMM
    }
  }
}

// ---------------------------------------------------------------------------
// host
// ---------------------------------------------------------------------------
extern "C" void kernel_launch(void* const* d_in, const int* in_sizes, int n_in,
                              void* d_out, int out_size, void* d_ws, size_t ws_size,
                              hipStream_t stream) {
  (void)in_sizes; (void)n_in; (void)out_size; (void)ws_size;

  const int*   inputs = (const int*)d_in[0];
  const float* hidden = (const float*)d_in[1];
  const float* emb    = (const float*)d_in[2];
  const float* Wr     = (const float*)d_in[3];
  const float* br     = (const float*)d_in[4];
  const float* Wz     = (const float*)d_in[5];
  const float* bz     = (const float*)d_in[6];
  const float* Wh     = (const float*)d_in[7];
  const float* bh     = (const float*)d_in[8];
  const float* Uz     = (const float*)d_in[9];
  const float* buz    = (const float*)d_in[10];
  const float* Uh     = (const float*)d_in[11];
  const float* buh    = (const float*)d_in[12];
  const float* Wy     = (const float*)d_in[13];
  const float* by     = (const float*)d_in[14];
  float* out = (float*)d_out;

  // workspace carve-up (256B aligned regions)
  char* ws = (char*)d_ws;
  size_t off = 0;
  auto take = [&](size_t bytes) -> char* {
    char* p = ws + off;
    off += (bytes + 255) & ~(size_t)255;
    return p;
  };
  __bf16* Wrb  = (__bf16*)take((size_t)2 * H * E * 2);
  __bf16* Wzb  = (__bf16*)take((size_t)2 * H * E * 2);
  __bf16* Whb  = (__bf16*)take((size_t)2 * H * E * 2);
  __bf16* Uzb  = (__bf16*)take((size_t)2 * H * H * 2);
  __bf16* Uhb  = (__bf16*)take((size_t)2 * H * H * 2);
  __bf16* Wyb  = (__bf16*)take((size_t)V * H * 2);
  __bf16* Xbf  = (__bf16*)take((size_t)SB * E * 2);
  float*  Ar   = (float*)take((size_t)SB * H * 4);
  float*  Az   = (float*)take((size_t)SB * H * 4);
  float*  Ah   = (float*)take((size_t)SB * H * 4);
  __bf16* H1bf = (__bf16*)take((size_t)SB * H * 2);
  __bf16* R0bf = (__bf16*)take((size_t)B * H * 2);
  __bf16* R1bf = (__bf16*)take((size_t)B * H * 2);
  float*  h0f[2]; __bf16* h0b[2]; float* h1f[2]; __bf16* h1b[2];
  for (int i = 0; i < 2; ++i) {
    h0f[i] = (float*)take((size_t)B * H * 4);
    h1f[i] = (float*)take((size_t)B * H * 4);
    h0b[i] = (__bf16*)take((size_t)B * H * 2);
    h1b[i] = (__bf16*)take((size_t)B * H * 2);
  }

  constexpr size_t STRIP = (size_t)16 * KD * 2;   // 32KB per 16-row A strip

  // 1) weights -> bf16
  cvt_kernel<<<2048, 256, 0, stream>>>(Wr, Wrb, (size_t)2 * H * E);
  cvt_kernel<<<2048, 256, 0, stream>>>(Wz, Wzb, (size_t)2 * H * E);
  cvt_kernel<<<2048, 256, 0, stream>>>(Wh, Whb, (size_t)2 * H * E);
  cvt_kernel<<<2048, 256, 0, stream>>>(Uz, Uzb, (size_t)2 * H * H);
  cvt_kernel<<<2048, 256, 0, stream>>>(Uh, Uhb, (size_t)2 * H * H);
  cvt_kernel<<<2048, 256, 0, stream>>>(Wy, Wyb, (size_t)V * H);

  // 2) embedding gather (+bf16)
  embed_kernel<<<SB, 256, 0, stream>>>(inputs, emb, Xbf);

  // 3) init hidden state ping-pong buffers
  inith_kernel<<<(B * H + 255) / 256, 256, 0, stream>>>(hidden, h0f[0], h0b[0], h1f[0], h1b[0]);

  // 4) precompute layer-0 input gates across ALL timesteps: A* = X*W*0^T + b*0
  {
    dim3 grid((H / 64 + 7) / 8, SB / 32);          // (2, 128), MT=2
    gemm_bias_kernel<2><<<grid, 256, 2 * STRIP, stream>>>(Xbf, Wrb, br, Ar, H, H / 64);
    gemm_bias_kernel<2><<<grid, 256, 2 * STRIP, stream>>>(Xbf, Wzb, bz, Az, H, H / 64);
    gemm_bias_kernel<2><<<grid, 256, 2 * STRIP, stream>>>(Xbf, Whb, bh, Ah, H, H / 64);
  }

  // layer-1 weight/bias slices
  const __bf16 *Wr1 = Wrb + (size_t)H * E, *Wz1 = Wzb + (size_t)H * E, *Wh1 = Whb + (size_t)H * E;
  const __bf16 *Uz0 = Uzb, *Uh0 = Uhb, *Uz1 = Uzb + (size_t)H * H, *Uh1 = Uhb + (size_t)H * H;
  const float *br1 = br + H, *bz1 = bz + H, *bh1 = bh + H;
  const float *buz0 = buz, *buh0 = buh, *buz1 = buz + H, *buh1 = buh + H;

  // 5) sequential recurrence
  dim3 sgrid(2, 4);                                 // 16 n-groups x 4 m-tiles
  for (int t = 0; t < S; ++t) {
    int cur = t & 1, nxt = cur ^ 1;
    const float* Ar_t = Ar + (size_t)t * B * H;
    const float* Az_t = Az + (size_t)t * B * H;
    const float* Ah_t = Ah + (size_t)t * B * H;

    r0_kernel<<<(B * H + 255) / 256, 256, 0, stream>>>(Ar_t, h0f[cur], R0bf);
    step0_kernel<<<sgrid, 256, 2 * STRIP, stream>>>(Az_t, Ah_t, h0f[cur], h0b[cur], R0bf,
                                                    Uz0, Uh0, buz0, buh0, h0f[nxt], h0b[nxt]);
    l1r_kernel<<<sgrid, 256, STRIP, stream>>>(h0b[nxt], Wr1, br1, h1f[cur], R1bf);
    step1_kernel<<<sgrid, 256, 3 * STRIP, stream>>>(h0b[nxt], h1b[cur], R1bf,
                                                    Wz1, Wh1, Uz1, Uh1,
                                                    bz1, buz1, bh1, buh1,
                                                    h1f[cur], h1f[nxt], h1b[nxt],
                                                    H1bf + (size_t)t * B * H);
  }

  // 6) logits = H1 * Wy^T + by  -> d_out[:S*B*V]
  {
    int nfull = V / 64;                             // 156 full 64-col groups
    dim3 grid((nfull + 7) / 8, SB / 32);            // (20, 128), MT=2
    gemm_bias_kernel<2><<<grid, 256, 2 * STRIP, stream>>>(H1bf, Wyb, by, out, V, nfull);
    // last 16 columns (9984..9999): one 16x16 tile per wave
    gemm_bias_tail_kernel<<<SB / 16 / 8, 256, 0, stream>>>(H1bf, Wyb, by, out, V, nfull * 64);
  }

  // 7) final hidden -> d_out tail (after S steps, state lives in buffer index 0)
  finalh_kernel<<<(B * H + 255) / 256, 256, 0, stream>>>(h0f[0], h1f[0],
                                                         out + (size_t)SB * V);
}
